// OpenProtTransformerBlock_48636209660016
// MI455X (gfx1250) — matrix operand order repaired and checked by reading, not compiled, not run on hardware
//
#include <hip/hip_runtime.h>
#include <hip/hip_bf16.h>
#include <math.h>

// ---------------------------------------------------------------------------
// Types for CDNA5 WMMA (gfx1250, wave32)
// ---------------------------------------------------------------------------
typedef __bf16 bf16t;
typedef __attribute__((ext_vector_type(16))) bf16t v16bf;
typedef __attribute__((ext_vector_type(8)))  float v8f;

#define LSEQ  512
#define DIMD  1024
#define NHEAD 16
#define HDIM  64
#define PDIM  128
#define FFD   4096              // FF * DIM
#define FFP   512               // FF * PD
#define NPAIR (LSEQ * LSEQ)     // 262144 pair positions

__device__ __forceinline__ int lane_id() { return (int)(threadIdx.x & 31u); }

__device__ __forceinline__ v8f wmma_bf16(v16bf a, v16bf b, v8f c) {
  // D = A(16x32 bf16) * B(32x16 bf16) + C(16x16 f32)
  return __builtin_amdgcn_wmma_f32_16x16x32_bf16(
      /*neg_a=*/false, a, /*neg_b=*/false, b,
      /*c_mod=*/(short)0, c, /*reuse_a=*/false, /*reuse_b=*/false);
}

__device__ __forceinline__ float sigmoidf_(float x) { return 1.f / (1.f + __expf(-x)); }

// ---------------------------------------------------------------------------
// Fragment loaders (wave32; A and B share this code's contiguous k order, not the ISA table order)
// A 16x32 bf16: lane m (<16) = row m, elems[0..15]=K k0..k0+15
//               lane m+16    = row m, elems[0..15]=K k0+16..k0+31
// B 32x16:      lane n (<16) = col n, elems[0..15]=K k0..k0+15 ; lane n+16 = col n, K k0+16..k0+31
// C/D 16x16 f32: lane l, vgpr v -> row = v + 8*(l>>4), col = l&15
// ---------------------------------------------------------------------------
__device__ __forceinline__ v16bf load_a_bf16(const bf16t* A, int lda, int m0, int k0) {
  int l = lane_id(); int r = l & 15; int hi = l >> 4;
  const bf16t* p = A + (size_t)(m0 + r) * lda + k0 + hi * 16;
  v16bf f;
#pragma unroll
  for (int e = 0; e < 8; ++e) f[e] = p[e];
#pragma unroll
  for (int e = 0; e < 8; ++e) f[8 + e] = p[8 + e];
  return f;
}

__device__ __forceinline__ v16bf load_a_f32(const float* A, int lda, int m0, int k0) {
  int l = lane_id(); int r = l & 15; int hi = l >> 4;
  const float* p = A + (size_t)(m0 + r) * lda + k0 + hi * 16;
  v16bf f;
#pragma unroll
  for (int e = 0; e < 8; ++e) f[e] = (bf16t)p[e];
#pragma unroll
  for (int e = 0; e < 8; ++e) f[8 + e] = (bf16t)p[8 + e];
  return f;
}

// B = Q^T where Q is row-major (N x K) bf16: lane n reads its row contiguously (fast path).
// Used both for "X @ W" with pre-transposed weights Wt (N x K) and for X @ Y^T GEMMs.
__device__ __forceinline__ v16bf load_bt_bf16(const bf16t* Q, int ldq, int n0, int k0) {
  int l = lane_id(); int c = l & 15; int hi = l >> 4;
  const bf16t* p = Q + (size_t)(n0 + c) * ldq + k0 + hi * 16;
  v16bf f;
#pragma unroll
  for (int e = 0; e < 16; ++e) f[e] = p[e];
  return f;
}

// ---------------------------------------------------------------------------
// Weight preprocessing: W (K x N, f32, row-major) -> Wt (N x K, bf16)
// Makes every WMMA B-fragment a contiguous 32-byte-per-lane load.
// ---------------------------------------------------------------------------
__global__ void wconv(const float* __restrict__ W, bf16t* __restrict__ Wt, int K, int N) {
  int idx = blockIdx.x * blockDim.x + threadIdx.x;
  if (idx >= N * K) return;
  int n = idx / K, k = idx - n * K;
  Wt[idx] = (bf16t)W[(size_t)k * N + n];
}

// ---------------------------------------------------------------------------
// LayerNorm over rows (f32 in -> bf16 out), one thread per row
// ---------------------------------------------------------------------------
__global__ void ln_rows(const float* __restrict__ in, const float* __restrict__ g,
                        const float* __restrict__ b, bf16t* __restrict__ out,
                        int rows, int n) {
  int r = blockIdx.x * blockDim.x + threadIdx.x;
  if (r >= rows) return;
  const float* p = in + (size_t)r * n;
  float s = 0.f, s2 = 0.f;
  for (int i = 0; i < n; ++i) { float v = p[i]; s += v; s2 += v * v; }
  float mean = s / n;
  float inv = rsqrtf(s2 / n - mean * mean + 1e-5f);
  bf16t* o = out + (size_t)r * n;
  for (int i = 0; i < n; ++i) o[i] = (bf16t)((p[i] - mean) * inv * g[i] + b[i]);
}

// ---------------------------------------------------------------------------
// Pair->sequence bias: bias[h][i][j] = LN(z[i,j,:]) @ p2s_w (128x16)
// ---------------------------------------------------------------------------
__global__ void p2s_bias(const float* __restrict__ z, const float* __restrict__ g,
                         const float* __restrict__ b, const float* __restrict__ w,
                         float* __restrict__ biasbuf) {
  int p = blockIdx.x * blockDim.x + threadIdx.x;  // 0..NPAIR-1
  if (p >= NPAIR) return;
  const float* zp = z + (size_t)p * PDIM;
  float s = 0.f, s2 = 0.f;
  for (int c = 0; c < PDIM; ++c) { float v = zp[c]; s += v; s2 += v * v; }
  float mean = s / PDIM;
  float inv = rsqrtf(s2 / PDIM - mean * mean + 1e-5f);
  float acc[NHEAD];
#pragma unroll
  for (int h = 0; h < NHEAD; ++h) acc[h] = 0.f;
  for (int c = 0; c < PDIM; ++c) {
    float v = (zp[c] - mean) * inv * g[c] + b[c];
#pragma unroll
    for (int h = 0; h < NHEAD; ++h) acc[h] += v * w[c * NHEAD + h];
  }
#pragma unroll
  for (int h = 0; h < NHEAD; ++h) biasbuf[(size_t)h * NPAIR + p] = acc[h];
}

// ---------------------------------------------------------------------------
// Generic GEMM, 16x64 tile per wave (4 accumulators, A reused 4x):
//   out = act(A(MxK bf16) @ Wt^T + bias) [+ resid]
// Wt is the pre-transposed (N x K) bf16 weight.  transb: store outb as (N x M).
// ---------------------------------------------------------------------------
__global__ void gemm_aw(const bf16t* __restrict__ A, const bf16t* __restrict__ Wt,
                        const float* __restrict__ bias, const float* __restrict__ resid,
                        float* __restrict__ outf, bf16t* __restrict__ outb,
                        int M, int N, int K, int relu, int transb) {
  int n0 = blockIdx.x * 64, m0 = blockIdx.y * 16;
  v8f zero = {};
  v8f accs[4] = {zero, zero, zero, zero};
  int l = lane_id(); int r = l & 15;
  for (int k0 = 0; k0 < K; k0 += 32) {
    if (k0 + 32 < K) {  // global_prefetch_b8 of next k-step (L2 warm)
      __builtin_prefetch(A + (size_t)(m0 + r) * K + k0 + 32, 0, 3);
      __builtin_prefetch(Wt + (size_t)(n0 + r) * K + k0 + 32, 0, 3);
    }
    v16bf a = load_a_bf16(A, K, m0, k0);
#pragma unroll
    for (int t = 0; t < 4; ++t)
      accs[t] = wmma_bf16(a, load_bt_bf16(Wt, K, n0 + 16 * t, k0), accs[t]);
  }
  int c = l & 15, hi = l >> 4;
#pragma unroll
  for (int t = 0; t < 4; ++t) {
    int n = n0 + 16 * t + c;
    float bv = bias ? bias[n] : 0.f;
#pragma unroll
    for (int v = 0; v < 8; ++v) {
      int m = m0 + v + 8 * hi;
      float x = accs[t][v] + bv;
      if (relu) x = fmaxf(x, 0.f);
      if (resid) x += resid[(size_t)m * N + n];
      if (outf) outf[(size_t)m * N + n] = x;
      if (outb) {
        if (transb) outb[(size_t)n * M + m] = (bf16t)x;
        else        outb[(size_t)m * N + n] = (bf16t)x;
      }
    }
  }
}

// ---------------------------------------------------------------------------
// Attention logits: logits[h][i][j] = q.k/8 + bias, mask -> -1e9
// ---------------------------------------------------------------------------
__global__ void attn_logits(const bf16t* __restrict__ qb, const bf16t* __restrict__ kb,
                            const float* __restrict__ biasbuf, const float* __restrict__ mask,
                            float* __restrict__ logits) {
  int j0 = blockIdx.x * 16, i0 = blockIdx.y * 16, h = blockIdx.z;
  v8f acc = {};
  for (int kk = 0; kk < HDIM; kk += 32) {
    v16bf a = load_a_bf16(qb, DIMD, i0, h * HDIM + kk);
    v16bf b = load_bt_bf16(kb, DIMD, j0, h * HDIM + kk);
    acc = wmma_bf16(a, b, acc);
  }
  int c = lane_id() & 15, hi = lane_id() >> 4;
  int j = j0 + c;
  float mj = mask[j];
  size_t plane = (size_t)h * NPAIR;
#pragma unroll
  for (int v = 0; v < 8; ++v) {
    int i = i0 + v + 8 * hi;
    float lg = acc[v] * 0.125f + biasbuf[plane + (size_t)i * LSEQ + j];
    if (mj <= 0.5f) lg = -1e9f;
    logits[plane + (size_t)i * LSEQ + j] = lg;
  }
}

__global__ void softmax_rows(float* __restrict__ logits) {
  int r = blockIdx.x * blockDim.x + threadIdx.x;  // 0..NHEAD*LSEQ-1
  if (r >= NHEAD * LSEQ) return;
  float* p = logits + (size_t)r * LSEQ;
  float mx = -3.4e38f;
  for (int j = 0; j < LSEQ; ++j) mx = fmaxf(mx, p[j]);
  float s = 0.f;
  for (int j = 0; j < LSEQ; ++j) { float e = __expf(p[j] - mx); p[j] = e; s += e; }
  float inv = 1.f / s;
  for (int j = 0; j < LSEQ; ++j) p[j] *= inv;
}

// o[i][h*64+n] = attn[h] @ v[h].  vbT is V stored transposed (1024 x 512) so B
// fragments are contiguous row reads.
__global__ void attn_v(const float* __restrict__ logits, const bf16t* __restrict__ vbT,
                       bf16t* __restrict__ ob) {
  int n0 = blockIdx.x * 16, i0 = blockIdx.y * 16, h = blockIdx.z;
  const float* Ap = logits + (size_t)h * NPAIR;
  v8f acc = {};
  for (int k0 = 0; k0 < LSEQ; k0 += 32) {
    v16bf a = load_a_f32(Ap, LSEQ, i0, k0);
    v16bf b = load_bt_bf16(vbT, LSEQ, h * HDIM + n0, k0);
    acc = wmma_bf16(a, b, acc);
  }
  int c = lane_id() & 15, hi = lane_id() >> 4;
  int n = h * HDIM + n0 + c;
#pragma unroll
  for (int v = 0; v < 8; ++v)
    ob[(size_t)(i0 + v + 8 * hi) * DIMD + n] = (bf16t)acc[v];
}

// ---------------------------------------------------------------------------
// s2p outer product: z_out = z_in + concat(qh[j]*kh[i], qh[j]-kh[i]) @ W + b
// ---------------------------------------------------------------------------
__global__ void s2p_outer(const float* __restrict__ spair, const bf16t* __restrict__ Wt,
                          const float* __restrict__ bias, const float* __restrict__ zin,
                          float* __restrict__ zout) {
  __shared__ bf16t tile[16 * PDIM];
  int p0 = blockIdx.x * 16;
  int lane = lane_id();
  if (lane < 16) {
    int p = p0 + lane; int i = p >> 9, j = p & 511;
    const float* qh = spair + (size_t)j * PDIM;       // qh = s[j, :64]
    const float* kh = spair + (size_t)i * PDIM + 64;  // kh = s[i, 64:]
    for (int c = 0; c < 64; ++c) tile[lane * PDIM + c] = (bf16t)(qh[c] * kh[c]);
    for (int c = 0; c < 64; ++c) tile[lane * PDIM + 64 + c] = (bf16t)(qh[c] - kh[c]);
  }
  __syncthreads();
  int c = lane & 15, hi = lane >> 4;
  for (int nt = 0; nt < 8; ++nt) {
    v8f acc = {};
    for (int k0 = 0; k0 < PDIM; k0 += 32) {
      v16bf a = load_a_bf16(tile, PDIM, 0, k0);
      v16bf b = load_bt_bf16(Wt, PDIM, nt * 16, k0);
      acc = wmma_bf16(a, b, acc);
    }
    int ch = nt * 16 + c;
    float bv = bias[ch];
#pragma unroll
    for (int v = 0; v < 8; ++v) {
      int p = p0 + v + 8 * hi;
      zout[(size_t)p * PDIM + ch] = zin[(size_t)p * PDIM + ch] + acc[v] + bv;
    }
  }
}

// ---------------------------------------------------------------------------
// Triangle stage 1: zl = LN(z); a = m*ap(zl)*sig(ag(zl)); b = m*bp*sig(bg);
// g = sig(g(zl)).  a,b scattered channel-major (c,row,col), flipped for incoming.
// ---------------------------------------------------------------------------
__global__ void tri_gates(const float* __restrict__ z, const float* __restrict__ lng,
                          const float* __restrict__ lnb,
                          const bf16t* __restrict__ Wap, const float* __restrict__ bap,
                          const bf16t* __restrict__ Wag, const float* __restrict__ bag,
                          const bf16t* __restrict__ Wbp, const float* __restrict__ bbp,
                          const bf16t* __restrict__ Wbg, const float* __restrict__ bbg,
                          const bf16t* __restrict__ Wg,  const float* __restrict__ bgg,
                          const float* __restrict__ mask,
                          bf16t* __restrict__ a_t, bf16t* __restrict__ b_t,
                          bf16t* __restrict__ gbuf, int flip) {
  __shared__ bf16t zl[16 * PDIM];
  __shared__ float mfac_s[16];
  int p0 = blockIdx.x * 16;
  int lane = lane_id();
  if (lane < 16) {
    int p = p0 + lane; int i = p >> 9, j = p & 511;
    const float* zp = z + (size_t)p * PDIM;
    float s = 0.f, s2 = 0.f;
    for (int c = 0; c < PDIM; ++c) { float v = zp[c]; s += v; s2 += v * v; }
    float mean = s / PDIM;
    float inv = rsqrtf(s2 / PDIM - mean * mean + 1e-5f);
    for (int c = 0; c < PDIM; ++c)
      zl[lane * PDIM + c] = (bf16t)((zp[c] - mean) * inv * lng[c] + lnb[c]);
    mfac_s[lane] = mask[i] * mask[j];
  }
  __syncthreads();
  int c = lane & 15, hi = lane >> 4;
  for (int nt = 0; nt < 8; ++nt) {
    int ch = nt * 16 + c;
    auto proj2_store = [&](const bf16t* Wp, const float* bp_, const bf16t* Wg2,
                           const float* bg2, bf16t* dst) {
      v8f accp = {}; v8f accg = {};
      for (int k0 = 0; k0 < PDIM; k0 += 32) {
        v16bf af = load_a_bf16(zl, PDIM, 0, k0);
        accp = wmma_bf16(af, load_bt_bf16(Wp, PDIM, nt * 16, k0), accp);
        accg = wmma_bf16(af, load_bt_bf16(Wg2, PDIM, nt * 16, k0), accg);
      }
      float bpv = bp_[ch], bgv = bg2[ch];
#pragma unroll
      for (int v = 0; v < 8; ++v) {
        int rr = v + 8 * hi; int p = p0 + rr;
        int i = p >> 9, j = p & 511;
        float val = mfac_s[rr] * (accp[v] + bpv) * sigmoidf_(accg[v] + bgv);
        size_t idx = (size_t)ch * NPAIR +
                     (flip ? ((size_t)j * LSEQ + i) : ((size_t)i * LSEQ + j));
        dst[idx] = (bf16t)val;
      }
    };
    proj2_store(Wap, bap, Wag, bag, a_t);
    proj2_store(Wbp, bbp, Wbg, bbg, b_t);
    v8f accG = {};
    for (int k0 = 0; k0 < PDIM; k0 += 32) {
      v16bf af = load_a_bf16(zl, PDIM, 0, k0);
      accG = wmma_bf16(af, load_bt_bf16(Wg, PDIM, nt * 16, k0), accG);
    }
    float bgv = bgg[ch];
#pragma unroll
    for (int v = 0; v < 8; ++v) {
      int p = p0 + v + 8 * hi;
      gbuf[(size_t)p * PDIM + ch] = (bf16t)sigmoidf_(accG[v] + bgv);
    }
  }
}

// ---------------------------------------------------------------------------
// Triangle stage 2: per-channel GEMM  xt[c] = A_c (512x512) @ B_c^T (512x512)
// 32x32 output per wave (2x2 blocking, each fragment reused twice).
// ---------------------------------------------------------------------------
__global__ void tri_einsum(const bf16t* __restrict__ a_t, const bf16t* __restrict__ b_t,
                           bf16t* __restrict__ xt) {
  int j0 = blockIdx.x * 32, i0 = blockIdx.y * 32, ch = blockIdx.z;
  const bf16t* A = a_t + (size_t)ch * NPAIR;
  const bf16t* Q = b_t + (size_t)ch * NPAIR;
  v8f zero = {};
  v8f acc00 = zero, acc01 = zero, acc10 = zero, acc11 = zero;
  for (int k0 = 0; k0 < LSEQ; k0 += 32) {
    v16bf a0 = load_a_bf16(A, LSEQ, i0, k0);
    v16bf a1 = load_a_bf16(A, LSEQ, i0 + 16, k0);
    v16bf b0 = load_bt_bf16(Q, LSEQ, j0, k0);
    v16bf b1 = load_bt_bf16(Q, LSEQ, j0 + 16, k0);
    acc00 = wmma_bf16(a0, b0, acc00);
    acc01 = wmma_bf16(a0, b1, acc01);
    acc10 = wmma_bf16(a1, b0, acc10);
    acc11 = wmma_bf16(a1, b1, acc11);
  }
  bf16t* X = xt + (size_t)ch * NPAIR;
  int c = lane_id() & 15, hi = lane_id() >> 4;
#pragma unroll
  for (int v = 0; v < 8; ++v) {
    int r0 = i0 + v + 8 * hi, r1 = r0 + 16;
    X[(size_t)r0 * LSEQ + j0 + c]      = (bf16t)acc00[v];
    X[(size_t)r0 * LSEQ + j0 + 16 + c] = (bf16t)acc01[v];
    X[(size_t)r1 * LSEQ + j0 + c]      = (bf16t)acc10[v];
    X[(size_t)r1 * LSEQ + j0 + 16 + c] = (bf16t)acc11[v];
  }
}

// ---------------------------------------------------------------------------
// Triangle stage 3: z += (LN_c(xt) @ z_w + z_b) * g
// ---------------------------------------------------------------------------
__global__ void tri_out(const bf16t* __restrict__ xt, const float* __restrict__ lng,
                        const float* __restrict__ lnb, const bf16t* __restrict__ Wz,
                        const float* __restrict__ bz, const bf16t* __restrict__ gbuf,
                        float* __restrict__ z) {
  __shared__ bf16t xl[16 * PDIM];
  int p0 = blockIdx.x * 16;
  int lane = lane_id();
  if (lane < 16) {
    int p = p0 + lane;
    float s = 0.f, s2 = 0.f;
    for (int cc = 0; cc < PDIM; ++cc) {
      float v = (float)xt[(size_t)cc * NPAIR + p]; s += v; s2 += v * v;
    }
    float mean = s / PDIM;
    float inv = rsqrtf(s2 / PDIM - mean * mean + 1e-5f);
    for (int cc = 0; cc < PDIM; ++cc) {
      float v = (float)xt[(size_t)cc * NPAIR + p];
      xl[lane * PDIM + cc] = (bf16t)((v - mean) * inv * lng[cc] + lnb[cc]);
    }
  }
  __syncthreads();
  int c = lane & 15, hi = lane >> 4;
  for (int nt = 0; nt < 8; ++nt) {
    v8f acc = {};
    for (int k0 = 0; k0 < PDIM; k0 += 32)
      acc = wmma_bf16(load_a_bf16(xl, PDIM, 0, k0),
                      load_bt_bf16(Wz, PDIM, nt * 16, k0), acc);
    int ch = nt * 16 + c;
    float bv = bz[ch];
#pragma unroll
    for (int v = 0; v < 8; ++v) {
      int p = p0 + v + 8 * hi;
      float g = (float)gbuf[(size_t)p * PDIM + ch];
      z[(size_t)p * PDIM + ch] += (acc[v] + bv) * g;
    }
  }
}

// ---------------------------------------------------------------------------
// Pair MLP (fused, hidden 512 stays in LDS): z += relu(z@W1+b1)@W2 + b2
// ---------------------------------------------------------------------------
__global__ void pair_mlp(float* __restrict__ z, const bf16t* __restrict__ W1t,
                         const float* __restrict__ b1, const bf16t* __restrict__ W2t,
                         const float* __restrict__ b2) {
  __shared__ bf16t zin[16 * PDIM];
  __shared__ bf16t hid[16 * FFP];
  int p0 = blockIdx.x * 16;
  int lane = lane_id();
  if (lane < 16) {
    int p = p0 + lane;
    for (int c = 0; c < PDIM; ++c)
      zin[lane * PDIM + c] = (bf16t)z[(size_t)p * PDIM + c];
  }
  __syncthreads();
  int c = lane & 15, hi = lane >> 4;
  for (int nt = 0; nt < FFP / 16; ++nt) {
    v8f acc = {};
    for (int k0 = 0; k0 < PDIM; k0 += 32)
      acc = wmma_bf16(load_a_bf16(zin, PDIM, 0, k0),
                      load_bt_bf16(W1t, PDIM, nt * 16, k0), acc);
    float bv = b1[nt * 16 + c];
#pragma unroll
    for (int v = 0; v < 8; ++v) {
      int rr = v + 8 * hi;
      hid[rr * FFP + nt * 16 + c] = (bf16t)fmaxf(acc[v] + bv, 0.f);
    }
  }
  __syncthreads();
  for (int nt = 0; nt < 8; ++nt) {
    v8f acc = {};
    for (int k0 = 0; k0 < FFP; k0 += 32)
      acc = wmma_bf16(load_a_bf16(hid, FFP, 0, k0),
                      load_bt_bf16(W2t, FFP, nt * 16, k0), acc);
    int ch = nt * 16 + c;
    float bv = b2[ch];
#pragma unroll
    for (int v = 0; v < 8; ++v) {
      int p = p0 + v + 8 * hi;
      z[(size_t)p * PDIM + ch] += acc[v] + bv;
    }
  }
}

// ---------------------------------------------------------------------------
// Pose update: u = x@ffu_w+b; trans += R@u[:3]; rots = rots @ AA(u[3:])^T
// ---------------------------------------------------------------------------
__global__ void pose_update(const float* __restrict__ x, const float* __restrict__ W,
                            const float* __restrict__ b, const float* __restrict__ rots_in,
                            const float* __restrict__ trans_in, float* __restrict__ rots_out,
                            float* __restrict__ trans_out) {
  int l = blockIdx.x * blockDim.x + threadIdx.x;
  if (l >= LSEQ) return;
  const float* xp = x + (size_t)l * DIMD;
  float u[6];
#pragma unroll
  for (int j = 0; j < 6; ++j) u[j] = b[j];
  for (int d = 0; d < DIMD; ++d) {
    float xv = xp[d];
#pragma unroll
    for (int j = 0; j < 6; ++j) u[j] += xv * W[d * 6 + j];
  }
  const float* R0 = rots_in + (size_t)l * 9;
#pragma unroll
  for (int i = 0; i < 3; ++i)
    trans_out[l * 3 + i] = trans_in[l * 3 + i] +
                           R0[i * 3 + 0] * u[0] + R0[i * 3 + 1] * u[1] + R0[i * 3 + 2] * u[2];
  float rx = u[3], ry = u[4], rz = u[5];
  float ang = sqrtf(rx * rx + ry * ry + rz * rz);
  float half = 0.5f * ang;
  float sfac = (ang < 1e-6f) ? (0.5f - ang * ang / 48.f) : (sinf(half) / ang);
  float qw = cosf(half), qx = rx * sfac, qy = ry * sfac, qz = rz * sfac;
  float two_s = 2.f / (qw * qw + qx * qx + qy * qy + qz * qz);
  float R[9] = {1.f - two_s * (qy * qy + qz * qz), two_s * (qx * qy - qz * qw), two_s * (qx * qz + qy * qw),
                two_s * (qx * qy + qz * qw), 1.f - two_s * (qx * qx + qz * qz), two_s * (qy * qz - qx * qw),
                two_s * (qx * qz - qy * qw), two_s * (qy * qz + qx * qw), 1.f - two_s * (qx * qx + qy * qy)};
#pragma unroll
  for (int i = 0; i < 3; ++i)
#pragma unroll
    for (int k = 0; k < 3; ++k)
      rots_out[l * 9 + i * 3 + k] =
          R0[i * 3 + 0] * R[k * 3 + 0] + R0[i * 3 + 1] * R[k * 3 + 1] + R0[i * 3 + 2] * R[k * 3 + 2];
}

// ---------------------------------------------------------------------------
// Host orchestration
// ---------------------------------------------------------------------------
enum {
  P_MHA_LN_G = 0, P_MHA_LN_B, P_FF_LN_G, P_FF_LN_B, P_S2P_LN_G, P_S2P_LN_B,
  P_P2S_LN_G, P_P2S_LN_B,
  P_WQ, P_WK, P_WV, P_WO, P_BQ, P_BK, P_BV, P_BO,
  P_FF_W1, P_FF_B1, P_FF_W2, P_FF_B2,
  P_S2P_PROJ_W, P_S2P_PROJ_B, P_S2P_O_W, P_S2P_O_B,
  P_P2S_W,
  P_TO_BASE,                    // lnin_g,lnin_b,lnout_g,lnout_b,ap_w,ap_b,ag_w,ag_b,
  P_TI_BASE = P_TO_BASE + 16,   // bp_w,bp_b,bg_w,bg_b,z_w,z_b,g_w,g_b
  P_MP_W1 = P_TI_BASE + 16, P_MP_B1, P_MP_W2, P_MP_B2,
  P_FFU_W, P_FFU_B
};
// offsets within a tri-param block
enum { T_LNIN_G = 0, T_LNIN_B, T_LNOUT_G, T_LNOUT_B, T_AP_W, T_AP_B, T_AG_W, T_AG_B,
       T_BP_W, T_BP_B, T_BG_W, T_BG_B, T_Z_W, T_Z_B, T_G_W, T_G_B };

extern "C" void kernel_launch(void* const* d_in, const int* in_sizes, int n_in,
                              void* d_out, int out_size, void* d_ws, size_t ws_size,
                              hipStream_t stream) {
  (void)in_sizes; (void)n_in; (void)out_size; (void)ws_size;
  const float* x_in     = (const float*)d_in[0];
  const float* z_in     = (const float*)d_in[1];
  const float* mask     = (const float*)d_in[2];
  const float* rots_in  = (const float*)d_in[3];
  const float* trans_in = (const float*)d_in[4];
  const float* const* P = (const float* const*)(d_in + 5);

  float* xo = (float*)d_out;                    // 512*1024
  float* zo = xo + (size_t)LSEQ * DIMD;         // 512*512*128
  float* ro = zo + (size_t)NPAIR * PDIM;        // 512*9
  float* to = ro + (size_t)LSEQ * 9;            // 512*3

  // workspace arena
  char* wsb = (char*)d_ws; size_t off = 0;
  auto carve = [&](size_t bytes) -> void* {
    void* p = wsb + off; off = (off + bytes + 255) & ~(size_t)255; return p;
  };
  bf16t* s_bf    = (bf16t*)carve((size_t)LSEQ * DIMD * 2);
  bf16t* qb      = (bf16t*)carve((size_t)LSEQ * DIMD * 2);
  bf16t* kb      = (bf16t*)carve((size_t)LSEQ * DIMD * 2);
  bf16t* vbT     = (bf16t*)carve((size_t)DIMD * LSEQ * 2);   // V transposed
  float* biasbuf = (float*)carve((size_t)NHEAD * NPAIR * 4);
  float* logits  = (float*)carve((size_t)NHEAD * NPAIR * 4);
  bf16t* ob      = (bf16t*)carve((size_t)LSEQ * DIMD * 2);
  float* x1      = (float*)carve((size_t)LSEQ * DIMD * 4);
  bf16t* h_bf    = (bf16t*)carve((size_t)LSEQ * DIMD * 2);
  bf16t* hid_bf  = (bf16t*)carve((size_t)LSEQ * FFD * 2);
  bf16t* sx_bf   = (bf16t*)carve((size_t)LSEQ * DIMD * 2);
  float* spair   = (float*)carve((size_t)LSEQ * PDIM * 4);
  bf16t* gbuf    = (bf16t*)carve((size_t)NPAIR * PDIM * 2);
  bf16t* a_t     = (bf16t*)carve((size_t)NPAIR * PDIM * 2);
  bf16t* b_t     = (bf16t*)carve((size_t)NPAIR * PDIM * 2);
  bf16t* xt      = (bf16t*)carve((size_t)NPAIR * PDIM * 2);

  // --- pre-transposed bf16 weights -------------------------------------
  auto wprep = [&](int pidx, int K, int N) -> bf16t* {
    bf16t* Wt = (bf16t*)carve((size_t)K * N * 2);
    int tot = K * N;
    wconv<<<(tot + 255) / 256, 256, 0, stream>>>(P[pidx], Wt, K, N);
    return Wt;
  };
  bf16t* wq_t  = wprep(P_WQ, DIMD, DIMD);
  bf16t* wk_t  = wprep(P_WK, DIMD, DIMD);
  bf16t* wv_t  = wprep(P_WV, DIMD, DIMD);
  bf16t* wo_t  = wprep(P_WO, DIMD, DIMD);
  bf16t* ff1_t = wprep(P_FF_W1, DIMD, FFD);
  bf16t* ff2_t = wprep(P_FF_W2, FFD, DIMD);
  bf16t* s2pp_t = wprep(P_S2P_PROJ_W, DIMD, PDIM);
  bf16t* s2po_t = wprep(P_S2P_O_W, PDIM, PDIM);
  bf16t* tri_t[2][6];
  for (int pass = 0; pass < 2; ++pass) {
    int base = (pass == 0 ? P_TO_BASE : P_TI_BASE);
    int widx[6] = {T_AP_W, T_AG_W, T_BP_W, T_BG_W, T_Z_W, T_G_W};
    for (int w = 0; w < 6; ++w) tri_t[pass][w] = wprep(base + widx[w], PDIM, PDIM);
  }
  bf16t* mp1_t = wprep(P_MP_W1, PDIM, FFP);
  bf16t* mp2_t = wprep(P_MP_W2, FFP, PDIM);

  const int NT = NPAIR / 16;  // 16384 pair tiles

  // 1) pair bias for attention (reads input z)
  p2s_bias<<<NPAIR / 256, 256, 0, stream>>>(z_in, P[P_P2S_LN_G], P[P_P2S_LN_B],
                                            P[P_P2S_W], biasbuf);
  // 2) MHA
  ln_rows<<<2, 256, 0, stream>>>(x_in, P[P_MHA_LN_G], P[P_MHA_LN_B], s_bf, LSEQ, DIMD);
  gemm_aw<<<dim3(16, 32), 32, 0, stream>>>(s_bf, wq_t, P[P_BQ], nullptr, nullptr, qb,
                                           LSEQ, DIMD, DIMD, 0, 0);
  gemm_aw<<<dim3(16, 32), 32, 0, stream>>>(s_bf, wk_t, P[P_BK], nullptr, nullptr, kb,
                                           LSEQ, DIMD, DIMD, 0, 0);
  gemm_aw<<<dim3(16, 32), 32, 0, stream>>>(s_bf, wv_t, P[P_BV], nullptr, nullptr, vbT,
                                           LSEQ, DIMD, DIMD, 0, 1);
  attn_logits<<<dim3(32, 32, NHEAD), 32, 0, stream>>>(qb, kb, biasbuf, mask, logits);
  softmax_rows<<<NHEAD * LSEQ / 256, 256, 0, stream>>>(logits);
  attn_v<<<dim3(4, 32, NHEAD), 32, 0, stream>>>(logits, vbT, ob);
  gemm_aw<<<dim3(16, 32), 32, 0, stream>>>(ob, wo_t, P[P_BO], x_in, x1, nullptr,
                                           LSEQ, DIMD, DIMD, 0, 0);
  // 3) FF
  ln_rows<<<2, 256, 0, stream>>>(x1, P[P_FF_LN_G], P[P_FF_LN_B], h_bf, LSEQ, DIMD);
  gemm_aw<<<dim3(64, 32), 32, 0, stream>>>(h_bf, ff1_t, P[P_FF_B1], nullptr, nullptr,
                                           hid_bf, LSEQ, FFD, DIMD, 1, 0);
  gemm_aw<<<dim3(16, 32), 32, 0, stream>>>(hid_bf, ff2_t, P[P_FF_B2], x1, xo, nullptr,
                                           LSEQ, DIMD, FFD, 0, 0);
  // 4) sequence -> pair
  ln_rows<<<2, 256, 0, stream>>>(xo, P[P_S2P_LN_G], P[P_S2P_LN_B], sx_bf, LSEQ, DIMD);
  gemm_aw<<<dim3(2, 32), 32, 0, stream>>>(sx_bf, s2pp_t, P[P_S2P_PROJ_B], nullptr,
                                          spair, nullptr, LSEQ, PDIM, DIMD, 0, 0);
  s2p_outer<<<NT, 32, 0, stream>>>(spair, s2po_t, P[P_S2P_O_B], z_in, zo);

  // 5) triangle multiplications (outgoing then incoming)
  for (int pass = 0; pass < 2; ++pass) {
    const float* const* T = P + (pass == 0 ? P_TO_BASE : P_TI_BASE);
    bf16t* const* W = tri_t[pass];
    int flip = pass;  // 0 = outgoing (ik x jk), 1 = incoming (ki x kj)
    tri_gates<<<NT, 32, 0, stream>>>(zo, T[T_LNIN_G], T[T_LNIN_B],
                                     W[0], T[T_AP_B], W[1], T[T_AG_B],
                                     W[2], T[T_BP_B], W[3], T[T_BG_B],
                                     W[5], T[T_G_B], mask, a_t, b_t, gbuf, flip);
    tri_einsum<<<dim3(16, 16, PDIM), 32, 0, stream>>>(a_t, b_t, xt);
    tri_out<<<NT, 32, 0, stream>>>(xt, T[T_LNOUT_G], T[T_LNOUT_B], W[4], T[T_Z_B],
                                   gbuf, zo);
  }

  // 6) pair MLP
  pair_mlp<<<NT, 32, 0, stream>>>(zo, mp1_t, P[P_MP_B1], mp2_t, P[P_MP_B2]);

  // 7) pose update
  pose_update<<<2, 256, 0, stream>>>(xo, P[P_FFU_W], P[P_FFU_B], rots_in, trans_in, ro, to);
}
